// MOELinear_38259568673108
// MI455X (gfx1250) — matrix operand order repaired and checked by reading, not compiled, not run on hardware
//
#include <hip/hip_runtime.h>
#include <hip/hip_bf16.h>

// ---------------- problem constants (match reference) ----------------
#define TOKENS      8192
#define D_MODEL     2048
#define NUM_EXPERTS 8
#define TOPK        2

// ---------------- vector types ----------------
typedef __attribute__((ext_vector_type(16))) __bf16   v16bf;
typedef __attribute__((ext_vector_type(8)))  __bf16   v8bf;
typedef __attribute__((ext_vector_type(8)))  float    v8f;
typedef __attribute__((ext_vector_type(4)))  float    v4f;
typedef __attribute__((ext_vector_type(4)))  unsigned v4u;
typedef __attribute__((ext_vector_type(4)))  int      v4i;

// address-space-qualified 16B pointers for the async copy builtin
// (clang prints AS1 as "__device__", AS3 as "__shared__" in HIP mode)
typedef __attribute__((address_space(1))) v4i* gptr_b128;
typedef __attribute__((address_space(3))) v4i* lptr_b128;

// pack two fp32 -> two bf16 (RNE) in one dword
__device__ __forceinline__ unsigned pk2bf(float a, float b) {
    unsigned ua = __builtin_bit_cast(unsigned, a);
    unsigned ub = __builtin_bit_cast(unsigned, b);
    ua += 0x7FFFu + ((ua >> 16) & 1u);
    ub += 0x7FFFu + ((ub >> 16) & 1u);
    return (ua >> 16) | (ub & 0xFFFF0000u);
}

__device__ __forceinline__ v16bf cat16(v8bf lo, v8bf hi) {
    v16bf r;
#pragma unroll
    for (int i = 0; i < 8; ++i) { r[i] = lo[i]; r[i + 8] = hi[i]; }
    return r;
}

// ---------------- gfx1250 async global->LDS copy (guarded) ----------------
#if __has_builtin(__builtin_amdgcn_global_load_async_to_lds_b128) && \
    __has_builtin(__builtin_amdgcn_s_wait_asynccnt)
#define USE_ASYNC_LDS 1
#else
#define USE_ASYNC_LDS 0
#endif

__device__ __forceinline__ void copy16B_g2l(const __bf16* g, __bf16* l) {
#if USE_ASYNC_LDS
    __builtin_amdgcn_global_load_async_to_lds_b128(
        (gptr_b128)const_cast<__bf16*>(g),
        (lptr_b128)l, 0, 0);
#else
    *(v8bf*)l = *(const v8bf*)g;
#endif
}

// =====================================================================
// Kernel 1: gate logits + top-2 + softmax + routing scatter.
// One wave32 per token; fixed-stride buckets (max/expert = TOKENS) so
// no prefix pass is needed and the capture graph stays static.
// =====================================================================
__global__ __launch_bounds__(256) void moe_gate_route(
    const float* __restrict__ x,
    const float* __restrict__ Wg,
    const float* __restrict__ bg,
    int*   __restrict__ counts,
    int*   __restrict__ btok,
    float* __restrict__ bw)
{
    const int wave = threadIdx.x >> 5;
    const int lane = threadIdx.x & 31;
    const int tok  = blockIdx.x * 8 + wave;

    const float* xr = x + (size_t)tok * D_MODEL;

    float acc[NUM_EXPERTS];
#pragma unroll
    for (int e = 0; e < NUM_EXPERTS; ++e) acc[e] = 0.0f;

    for (int k = lane; k < D_MODEL; k += 32) {
        const float xv = xr[k];
#pragma unroll
        for (int e = 0; e < NUM_EXPERTS; ++e)
            acc[e] = fmaf(xv, Wg[e * D_MODEL + k], acc[e]);
    }
#pragma unroll
    for (int e = 0; e < NUM_EXPERTS; ++e) {
#pragma unroll
        for (int off = 16; off > 0; off >>= 1)
            acc[e] += __shfl_xor(acc[e], off, 32);
    }

    if (lane == 0) {
        float lg[NUM_EXPERTS];
#pragma unroll
        for (int e = 0; e < NUM_EXPERTS; ++e) lg[e] = acc[e] + bg[e];
        int i0 = 0;
#pragma unroll
        for (int e = 1; e < NUM_EXPERTS; ++e) if (lg[e] > lg[i0]) i0 = e;
        int i1 = (i0 == 0) ? 1 : 0;
#pragma unroll
        for (int e = 0; e < NUM_EXPERTS; ++e)
            if (e != i0 && lg[e] > lg[i1]) i1 = e;

        const float l0 = lg[i0], l1 = lg[i1];
        const float m  = fmaxf(l0, l1);
        const float e0 = __expf(l0 - m), e1 = __expf(l1 - m);
        const float inv = 1.0f / (e0 + e1);

        int p0 = atomicAdd(&counts[i0], 1);
        btok[i0 * TOKENS + p0] = tok;
        bw  [i0 * TOKENS + p0] = e0 * inv;
        int p1 = atomicAdd(&counts[i1], 1);
        btok[i1 * TOKENS + p1] = tok;
        bw  [i1 * TOKENS + p1] = e1 * inv;
    }
}

// =====================================================================
// Kernel 2: x (fp32) -> xb (bf16), 8 elements/thread, 16B stores.
// =====================================================================
__global__ __launch_bounds__(256) void convert_x_bf16(
    const float* __restrict__ x, __bf16* __restrict__ xb)
{
    const size_t i = ((size_t)blockIdx.x * 256 + threadIdx.x) * 8;
    v4f a = *(const v4f*)(x + i);
    v4f b = *(const v4f*)(x + i + 4);
    v4u p = { pk2bf(a.x, a.y), pk2bf(a.z, a.w), pk2bf(b.x, b.y), pk2bf(b.z, b.w) };
    *(v4u*)(xb + i) = p;
}

// =====================================================================
// Kernel 3: We[e][n][k] (fp32) -> WeT[e][k][n] (bf16) via LDS tile
// transpose (64x64 tile / block). Makes GEMM's B staging a contiguous
// 16B-aligned copy in both global and LDS.
// =====================================================================
#define TT 64
__global__ __launch_bounds__(256) void transpose_We_bf16(
    const float* __restrict__ We, __bf16* __restrict__ WeT)
{
    const int e  = blockIdx.z;
    const int n0 = blockIdx.x * TT;
    const int k0 = blockIdx.y * TT;

    __shared__ float sT[TT][TT + 1];
    const int tid = threadIdx.x;

    { // load 64(n) x 64(k) floats, coalesced along k
        const int kq = (tid & 15) * 4;
        const int nr = tid >> 4;
#pragma unroll
        for (int i = 0; i < 4; ++i) {
            const int n = nr + i * 16;
            v4f v = *(const v4f*)(We + ((size_t)e * D_MODEL + n0 + n) * D_MODEL + k0 + kq);
            sT[n][kq + 0] = v.x;
            sT[n][kq + 1] = v.y;
            sT[n][kq + 2] = v.z;
            sT[n][kq + 3] = v.w;
        }
    }
    __syncthreads();
    { // write rows of k with n contiguous, packed bf16
        const int k  = tid >> 2;
        const int ns = (tid & 3) * 16;
        unsigned p[8];
#pragma unroll
        for (int j = 0; j < 8; ++j)
            p[j] = pk2bf(sT[ns + 2 * j][k], sT[ns + 2 * j + 1][k]);
        __bf16* dst = WeT + ((size_t)e * D_MODEL + (k0 + k)) * D_MODEL + n0 + ns;
        *(v4u*)(dst)     = (v4u){p[0], p[1], p[2], p[3]};
        *(v4u*)(dst + 8) = (v4u){p[4], p[5], p[6], p[7]};
    }
}

// =====================================================================
// Kernel 4: routed expert GEMM, all-bf16 operands from workspace.
// Block = 256 threads = 8 waves (4M x 2N); tile 128(M) x 128(N), K=32
// per step; each wave: 2x4 v_wmma_f32_16x16x32_bf16 per K-step.
//
// Double-buffered LDS + pipelined global_load_async_to_lds_b128:
// next tile's copies are issued before computing the current tile,
// then s_wait_asynccnt(4) retires only the current buffer's 4 ops
// (ASYNCcnt retires in order), overlapping DMA with WMMA.
// =====================================================================
#define BM 128
#define BN 128
#define BK 32
#define XPAD 8   // sX row = 40 bf16 = 80B : 16B aligned, conflict-free b128
#define WPAD 8   // sW row = 136 bf16 = 272B: 16B aligned, conflict-free b128

__global__ __launch_bounds__(256) void moe_expert_gemm(
    const __bf16* __restrict__ xb,    // [TOKENS, D_MODEL] bf16
    const __bf16* __restrict__ zrow,  // zero row for gather tails
    const __bf16* __restrict__ WeT,   // [E, K, N] bf16 (pre-transposed)
    const float*  __restrict__ be,    // [E, N] fp32
    const int*    __restrict__ counts,
    const int*    __restrict__ btok,
    const float*  __restrict__ bw,
    float* __restrict__ out)          // [TOKENS, D_MODEL] fp32, atomics
{
    const int expert = blockIdx.x >> 6;   // 64 worst-case m-tiles / expert
    const int mtile  = blockIdx.x & 63;
    const int nbase  = blockIdx.y * BN;

    const int cnt = counts[expert];
    const int m0  = mtile * BM;
    if (m0 >= cnt) return;                // uniform: EXEC stays all-1 for WMMA

    __shared__ alignas(16) __bf16 sX[2][BM][BK + XPAD];
    __shared__ alignas(16) __bf16 sW[2][BK][BN + WPAD];
    __shared__ int   sTok[BM];
    __shared__ float sWt[BM];

    const int tid  = threadIdx.x;
    const int wave = tid >> 5;
    const int lane = tid & 31;
    const int wy   = wave >> 1;           // 0..3 (M)
    const int wx   = wave & 1;            // 0..1 (N)
    const int l    = lane & 15;
    const int h    = lane >> 4;

    if (tid < BM) {
        const int idx = m0 + tid;
        if (idx < cnt) {
            sTok[tid] = btok[expert * TOKENS + idx];
            sWt [tid] = bw  [expert * TOKENS + idx];
        } else {
            sTok[tid] = -1;
            sWt [tid] = 0.0f;
        }
    }
    __syncthreads();

    // per-thread staging assignments (pure 16B copies, 4 per thread/tile)
    const int xrow = tid >> 1;            // 0..127
    const int xks  = (tid & 1) * 16;      // 0 / 16
    const int xt   = sTok[xrow];
    const __bf16* xsrc_base = (xt >= 0) ? (xb + (size_t)xt * D_MODEL + xks)
                                        : (zrow + xks);
    const int wk   = tid >> 3;            // 0..31
    const int wns  = (tid & 7) * 16;      // 0..112
    const __bf16* wsrc_base =
        WeT + ((size_t)expert * D_MODEL + wk) * D_MODEL + nbase + wns;

    auto stage = [&](int buf, int k0) {
        const __bf16* xs = xsrc_base + ((xt >= 0) ? k0 : 0);
        copy16B_g2l(xs,     &sX[buf][xrow][xks]);
        copy16B_g2l(xs + 8, &sX[buf][xrow][xks + 8]);
        const __bf16* ws = wsrc_base + (size_t)k0 * D_MODEL;
        copy16B_g2l(ws,     &sW[buf][wk][wns]);
        copy16B_g2l(ws + 8, &sW[buf][wk][wns + 8]);
    };

    v8f acc[2][4] = {};

    stage(0, 0);
    int buf = 0;
    for (int k0 = 0; k0 < D_MODEL; k0 += BK) {
        // prefetch next tile into the other buffer, then retire current
        if (k0 + BK < D_MODEL) {
            stage(buf ^ 1, k0 + BK);
#if USE_ASYNC_LDS
            __builtin_amdgcn_s_wait_asynccnt(4);  // 4 newest = prefetch
#endif
        } else {
#if USE_ASYNC_LDS
            __builtin_amdgcn_s_wait_asynccnt(0);
#endif
        }
        __syncthreads();

        // ---- fragments per wave32 VGPR layouts ----
        // A 16x32 bf16: lane l = row M; half h selects K groups:
        //   V0..3 = K[h*8..h*8+7], V4..7 = K[16+h*8..16+h*8+7]
        v16bf afrag[2], bfrag[4];
#pragma unroll
        for (int mi = 0; mi < 2; ++mi) {
            const int row = wy * 32 + mi * 16 + l;
            afrag[mi] = cat16(*(const v8bf*)&sX[buf][row][h * 8],
                              *(const v8bf*)&sX[buf][row][16 + h * 8]);
        }
        // B 32x16 bf16: lane l = col N; same K groups as A
#pragma unroll
        for (int ni = 0; ni < 4; ++ni) {
            const int nn = wx * 64 + ni * 16;
#pragma unroll
            for (int s = 0; s < 8; ++s) {
                bfrag[ni][s]     = sW[buf][h * 8 + s][nn + l];
                bfrag[ni][s + 8] = sW[buf][16 + h * 8 + s][nn + l];
            }
        }

#pragma unroll
        for (int mi = 0; mi < 2; ++mi)
#pragma unroll
            for (int ni = 0; ni < 4; ++ni)
                acc[mi][ni] = __builtin_amdgcn_wmma_f32_16x16x32_bf16(
                    false, afrag[mi], false, bfrag[ni],
                    (short)0, acc[mi][ni], false, false);

        __syncthreads();   // all reads of buf done before it is re-staged
        buf ^= 1;
    }

    // ---- epilogue: C/D layout lane = N, VGPR r -> M = r + 8*laneHalf ----
    // Row-major walk: one (token, weight) LDS read per output row,
    // then 4 global_atomic_add_f32 along N.
    float bias[4];
#pragma unroll
    for (int ni = 0; ni < 4; ++ni)
        bias[ni] = be[(size_t)expert * D_MODEL + nbase + wx * 64 + ni * 16 + l];

#pragma unroll
    for (int mi = 0; mi < 2; ++mi) {
#pragma unroll
        for (int r = 0; r < 8; ++r) {
            const int mrow = wy * 32 + mi * 16 + r + 8 * h;
            const int t  = sTok[mrow];
            const float wt = sWt[mrow];
            if (t >= 0) {
                float* orow = out + (size_t)t * D_MODEL + nbase + wx * 64 + l;
#pragma unroll
                for (int ni = 0; ni < 4; ++ni)
                    atomicAdd(orow + ni * 16, wt * (acc[mi][ni][r] + bias[ni]));
            }
        }
    }
}

// =====================================================================
// Launcher (graph-capture safe: memsetAsync + kernels only)
//
// ws layout (bytes):
//   0                : counts       (32)
//   256              : btok         (8*8192*4 = 256K)
//   256+256K         : bw           (256K)
//   OFF_XB           : xb  bf16     (32M)
//   OFF_ZROW         : zero row     (4K)
//   OFF_WET          : WeT bf16     (64M)      -> total ~96.6 MB
// =====================================================================
#define OFF_BTOK  ((size_t)256)
#define OFF_BW    (OFF_BTOK + (size_t)NUM_EXPERTS * TOKENS * 4)
#define OFF_XB    (OFF_BW   + (size_t)NUM_EXPERTS * TOKENS * 4)
#define OFF_ZROW  (OFF_XB   + (size_t)TOKENS * D_MODEL * 2)
#define OFF_WET   (OFF_ZROW + (size_t)4096)

extern "C" void kernel_launch(void* const* d_in, const int* in_sizes, int n_in,
                              void* d_out, int out_size, void* d_ws, size_t ws_size,
                              hipStream_t stream) {
    (void)in_sizes; (void)n_in; (void)out_size; (void)ws_size;

    const float* x  = (const float*)d_in[0];
    const float* Wg = (const float*)d_in[1];
    const float* bg = (const float*)d_in[2];
    const float* We = (const float*)d_in[3];
    const float* be = (const float*)d_in[4];
    float* out = (float*)d_out;

    char*   ws     = (char*)d_ws;
    int*    counts = (int*)ws;
    int*    btok   = (int*)(ws + OFF_BTOK);
    float*  bw     = (float*)(ws + OFF_BW);
    __bf16* xb     = (__bf16*)(ws + OFF_XB);
    __bf16* zrow   = (__bf16*)(ws + OFF_ZROW);
    __bf16* WeT    = (__bf16*)(ws + OFF_WET);

    (void)hipMemsetAsync(counts, 0, NUM_EXPERTS * sizeof(int), stream);
    (void)hipMemsetAsync(zrow, 0, 4096, stream);
    (void)hipMemsetAsync(out, 0, (size_t)TOKENS * D_MODEL * sizeof(float), stream);

    // routing
    moe_gate_route<<<TOKENS / 8, 256, 0, stream>>>(x, Wg, bg, counts, btok, bw);
    // precision pre-passes (bandwidth-bound, ~12us total at 23.3 TB/s)
    convert_x_bf16<<<(TOKENS * (size_t)D_MODEL) / (256 * 8), 256, 0, stream>>>(x, xb);
    {
        dim3 g(D_MODEL / TT, D_MODEL / TT, NUM_EXPERTS);
        transpose_We_bf16<<<g, 256, 0, stream>>>(We, WeT);
    }
    // routed matrix phase
    {
        dim3 g(NUM_EXPERTS * (TOKENS / BM), D_MODEL / BN);
        moe_expert_gemm<<<g, 256, 0, stream>>>(xb, zrow, WeT, be,
                                               counts, btok, bw, out);
    }
}
